// HINormer_40432822125221
// MI455X (gfx1250) — matrix operand order repaired and checked by reading, not compiled, not run on hardware
//
#include <hip/hip_runtime.h>
#include <hip/hip_fp16.h>

#define B_   4
#define N_   2048
#define D_   128
#define H_   4
#define HD_  32
#define RL_  4
#define R_   (B_ * N_)       // 8192 rows
#define NCH_ (N_ / 32)       // 64 key chunks of 32
#define LN_EPS 1e-5f

typedef __attribute__((ext_vector_type(16))) _Float16 v16h;
typedef __attribute__((ext_vector_type(8)))  float    v8f;
typedef __attribute__((ext_vector_type(2)))  float    v2f;

// ---------- cross-lane reductions within 16-lane halves (wave32) ----------
__device__ __forceinline__ float redSum16(float x) {
  x += __shfl_xor(x, 1, 32);
  x += __shfl_xor(x, 2, 32);
  x += __shfl_xor(x, 4, 32);
  x += __shfl_xor(x, 8, 32);
  return x;
}
__device__ __forceinline__ float redMax16(float x) {
  x = fmaxf(x, __shfl_xor(x, 1, 32));
  x = fmaxf(x, __shfl_xor(x, 2, 32));
  x = fmaxf(x, __shfl_xor(x, 4, 32));
  x = fmaxf(x, __shfl_xor(x, 8, 32));
  return x;
}

// A-fragment K index for 16-bit 16x32 A tile: VGPR v holds pair (K, K+1)
__device__ __forceinline__ int pairK(int v, int hi) {
  return ((v & 4) << 2) + 2 * (v & 3) + 8 * hi;
}

// ---------------------------- prep kernels ----------------------------
__global__ void prep_h_kernel(const float* __restrict__ h, _Float16* __restrict__ hh, int n) {
  for (int i = blockIdx.x * blockDim.x + threadIdx.x; i < n; i += gridDim.x * blockDim.x)
    hh[i] = (_Float16)h[i];
}

// Pre-swizzle W (row-major [out,in]) into per-lane WMMA B fragments of W^T.
__global__ void prep_w_kernel(const float* __restrict__ W, _Float16* __restrict__ WT) {
  int t = blockIdx.x * blockDim.x + threadIdx.x;
  if (t >= D_ * D_) return;
  int tile = t >> 9;          // 512 halves per 32x16 tile
  int lane = (t >> 4) & 31;
  int jj   = t & 15;
  int kb = tile >> 3, nb = tile & 7;
  int K    = kb * 32 + 8 * (lane >> 4) + (jj & 7) + 16 * (jj >> 3);
  int ncol = nb * 16 + (lane & 15);
  WT[t] = (_Float16)W[ncol * D_ + K];
}

__global__ void prep_r_kernel(const float* __restrict__ rh,
                              const float* __restrict__ Wrs,
                              const float* __restrict__ Wrt,
                              float* __restrict__ rk, float* __restrict__ rq) {
  int idx = blockIdx.x * blockDim.x + threadIdx.x;
  if (idx >= B_ * H_ * N_ * RL_) return;
  int r = idx & (RL_ - 1);
  int tmp = idx >> 2;
  int n = tmp % N_; tmp /= N_;
  int head = tmp % H_; int b = tmp / H_;
  const float* rrow = rh + (size_t)(b * N_ + n) * RL_;
  const float* ws = Wrs + (head * RL_ + r) * RL_;
  const float* wt = Wrt + (head * RL_ + r) * RL_;
  float sk = 0.f, sq = 0.f;
#pragma unroll
  for (int j = 0; j < RL_; ++j) { sk += ws[j] * rrow[j]; sq += wt[j] * rrow[j]; }
  rk[idx] = sk; rq[idx] = sq;
}

__global__ void zero_kernel(float* __restrict__ p, int n) {
  for (int i = blockIdx.x * blockDim.x + threadIdx.x; i < n; i += gridDim.x * blockDim.x)
    p[i] = 0.f;
}

// ------------------------- projection (WMMA f16) -------------------------
// One wave per 16x16 output tile of f = h @ W^T.  Reduces leaky(f)·attw into
// sbuf[(b*H+head)*N+n].  Wr path additionally scatters f directly into the
// V-fragment-swizzled layout consumed by the attention PV WMMAs:
//   vswz[(((b*H+head)*NCH + kc)<<10) + sub*512 + lane'*16 + j],
//   j = K_local & 15, lane' = (K_local>>4)*16 + n16   (K_local = key & 31)
__global__ void proj_kernel(const _Float16* __restrict__ hh,
                            const _Float16* __restrict__ WT,
                            const float* __restrict__ attw,
                            float* __restrict__ sbuf,
                            _Float16* __restrict__ vswz, int storeF) {
  int lane = threadIdx.x;
  int m = lane & 15, hi = lane >> 4;
  int rowBase = blockIdx.x * 16;
  int ct = blockIdx.y;

  v8f c = {};
#pragma unroll
  for (int kb = 0; kb < 4; ++kb) {
    const _Float16* arow = hh + (size_t)(rowBase + m) * D_ + kb * 32;
    v16h a;
#pragma unroll
    for (int v = 0; v < 8; ++v) {
      int k = pairK(v, hi);
      a[2 * v] = arow[k]; a[2 * v + 1] = arow[k + 1];
    }
    v16h b = *(const v16h*)(WT + (size_t)(kb * 8 + ct) * 512 + lane * 16);
    c = __builtin_amdgcn_wmma_f32_16x16x32_f16(false, a, false, b, (short)0, c, false, false);
  }

  int head = ct >> 1;
  int sub  = ct & 1;
  int bb = rowBase / N_;
  int n0 = rowBase % N_;
  float aw = attw[sub * 16 + m];
  float part[8];
#pragma unroll
  for (int v = 0; v < 8; ++v) {
    float x = c[v];                        // row = rowBase+v+8hi, col-in-16 = m
    if (storeF) {
      int key   = n0 + v + 8 * hi;
      int kc    = key >> 5;
      int Kl    = key & 31;
      int lane2 = ((Kl >> 4) << 4) + m;
      size_t idx = ((((size_t)bb * H_ + head) * NCH_ + kc) << 10) +
                   sub * 512 + lane2 * 16 + (Kl & 15);
      vswz[idx] = (_Float16)x;
    }
    float lx = x > 0.f ? x : 0.01f * x;    // leaky
    part[v] = redSum16(lx * aw);
  }
  if (m == 0) {
#pragma unroll
    for (int v = 0; v < 8; ++v) {
      int n = n0 + v + 8 * hi;
      atomicAdd(&sbuf[(size_t)(bb * H_ + head) * N_ + n], part[v]);
    }
  }
}

// ---------------- flash attention ----------------
// scores: v_wmma_f32_16x16x4_f32 (rank-4 relational bias) + sl/sr broadcasts
// PV:     v_wmma_f32_16x16x32_f16, V tiles double-buffered into LDS with
//         global_load_async_to_lds_b128 / s_wait_asynccnt
__global__ void attn_kernel(const float* __restrict__ sl, const float* __restrict__ sr,
                            const float* __restrict__ rk, const float* __restrict__ rq,
                            const _Float16* __restrict__ vswz, _Float16* __restrict__ hsa16) {
  __shared__ _Float16 sP[4][16 * 34];                 // P staging (C-layout -> A-frag)
  __shared__ __align__(16) char sV[4][2][2048];       // per-wave V double buffer
  int lane = threadIdx.x & 31;
  int wv = threadIdx.x >> 5;
  int m = lane & 15, hi = lane >> 4;
  int head = blockIdx.y, bz = blockIdx.z;
  int rowBase = (blockIdx.x * 4 + wv) * 16;
  size_t sb = ((size_t)bz * H_ + head) * N_;
  size_t vb0 = ((size_t)bz * H_ + head) * NCH_;       // chunk-tile base for this (b,head)

  // rank-4 A fragment (f32 16x4): a[v] = rk[row][v + 2*hi]
  v2f aR;
  { const float* p = rk + (sb + rowBase + m) * RL_ + 2 * hi; aR[0] = p[0]; aR[1] = p[1]; }
  float slr[8];
#pragma unroll
  for (int v = 0; v < 8; ++v) slr[v] = sl[sb + rowBase + v + 8 * hi];

  float mrun[8], lrun[8];
#pragma unroll
  for (int v = 0; v < 8; ++v) { mrun[v] = -1e30f; lrun[v] = 0.f; }
  v8f O0 = {}, O1 = {};
  _Float16* sp = sP[wv];

  unsigned ldsV[2];
  ldsV[0] = (unsigned)(uintptr_t)&sV[wv][0][0] + lane * 16;
  ldsV[1] = (unsigned)(uintptr_t)&sV[wv][1][0] + lane * 16;

  // async-copy one 2KB V chunk tile (32 keys x 32 cols, fragment-swizzled)
  auto issueV = [&](int kc1, int buf) {
    const char* gb = (const char*)(vswz + ((vb0 + kc1) << 10)) + lane * 16;
    unsigned lb = ldsV[buf];
    asm volatile("global_load_async_to_lds_b128 %0, %1, off"             :: "v"(lb), "v"(gb) : "memory");
    asm volatile("global_load_async_to_lds_b128 %0, %1, off offset:512"  :: "v"(lb), "v"(gb) : "memory");
    asm volatile("global_load_async_to_lds_b128 %0, %1, off offset:1024" :: "v"(lb), "v"(gb) : "memory");
    asm volatile("global_load_async_to_lds_b128 %0, %1, off offset:1536" :: "v"(lb), "v"(gb) : "memory");
  };

  issueV(0, 0);                                       // prologue: chunk 0

  for (int kc = 0; kc < NCH_; ++kc) {
    // previous iteration's LDS reads must land before we overwrite that buffer
    asm volatile("s_wait_dscnt 0x0" ::: "memory");
    if (kc + 1 < NCH_) issueV(kc + 1, (kc + 1) & 1);  // prefetch next chunk

    int key0 = kc * 32;
    float p[16];
#pragma unroll
    for (int sub = 0; sub < 2; ++sub) {
      int j0 = key0 + sub * 16;
      v2f bR;                                         // B frag (f32 4x16): rq[key][v+2hi]
      { const float* q = rq + (sb + j0 + m) * RL_ + 2 * hi; bR[0] = q[0]; bR[1] = q[1]; }
      v8f s = {};
      s = __builtin_amdgcn_wmma_f32_16x16x4_f32(false, aR, false, bR, (short)0, s, false, false);
      float srn = sr[sb + j0 + m];
#pragma unroll
      for (int v = 0; v < 8; ++v) p[sub * 8 + v] = s[v] + slr[v] + srn;
    }
    // online softmax over this 32-key chunk
    float scale[8];
#pragma unroll
    for (int v = 0; v < 8; ++v) {
      float mx = redMax16(fmaxf(p[v], p[8 + v]));
      float mn = fmaxf(mrun[v], mx);
      scale[v] = __expf(mrun[v] - mn);
      mrun[v] = mn;
    }
#pragma unroll
    for (int v = 0; v < 8; ++v) {
      p[v]     = __expf(p[v]     - mrun[v]);
      p[8 + v] = __expf(p[8 + v] - mrun[v]);
      float rs = redSum16(p[v] + p[8 + v]);
      lrun[v] = lrun[v] * scale[v] + rs;
      O0[v] *= scale[v]; O1[v] *= scale[v];
    }
    // wait for this chunk's async V tile to be resident in LDS
    if (kc + 1 < NCH_) asm volatile("s_wait_asynccnt 0x4" ::: "memory");
    else               asm volatile("s_wait_asynccnt 0x0" ::: "memory");

    // C-layout P -> LDS -> A fragment
#pragma unroll
    for (int v = 0; v < 8; ++v) {
      sp[(v + 8 * hi) * 34 + m]      = (_Float16)p[v];
      sp[(v + 8 * hi) * 34 + 16 + m] = (_Float16)p[8 + v];
    }
    asm volatile("s_wait_dscnt 0x0" ::: "memory");
    v16h pa;
#pragma unroll
    for (int v = 0; v < 8; ++v) {
      int k = 16 * hi + 2 * v;
      pa[2 * v] = sp[m * 34 + k]; pa[2 * v + 1] = sp[m * 34 + k + 1];
    }
    asm volatile("" ::: "memory");

    // V fragments straight out of LDS (contiguous 32B per lane)
    const _Float16* vtile = (const _Float16*)&sV[wv][kc & 1][0];
    v16h bv0 = *(const v16h*)(vtile + lane * 16);
    v16h bv1 = *(const v16h*)(vtile + 512 + lane * 16);
    O0 = __builtin_amdgcn_wmma_f32_16x16x32_f16(false, pa, false, bv0, (short)0, O0, false, false);
    O1 = __builtin_amdgcn_wmma_f32_16x16x32_f16(false, pa, false, bv1, (short)0, O1, false, false);
  }
  // normalize and store context into [B*N, D] f16 (cols head*32 .. head*32+31)
#pragma unroll
  for (int v = 0; v < 8; ++v) {
    float inv = 1.f / lrun[v];
    size_t row = (size_t)bz * N_ + rowBase + v + 8 * hi;
    hsa16[row * D_ + head * HD_ + m]      = (_Float16)(O0[v] * inv);
    hsa16[row * D_ + head * HD_ + 16 + m] = (_Float16)(O1[v] * inv);
  }
}

// -------- final: fh = h_sa @ Wf^T (f16 WMMA), residual + LayerNorm --------
__global__ void final_kernel(const _Float16* __restrict__ hsa16,
                             const _Float16* __restrict__ WfT,
                             const float* __restrict__ h,
                             const float* __restrict__ lng,
                             const float* __restrict__ lnb,
                             float* __restrict__ out) {
  int lane = threadIdx.x;
  int m = lane & 15, hi = lane >> 4;
  int rowBase = blockIdx.x * 16;

  // hoist the 4 A fragments (K blocks) once; reuse across all 8 col tiles
  v16h aF[4];
#pragma unroll
  for (int kb = 0; kb < 4; ++kb) {
    const _Float16* arow = hsa16 + (size_t)(rowBase + m) * D_ + kb * 32;
#pragma unroll
    for (int v = 0; v < 8; ++v) {
      int k = pairK(v, hi);
      aF[kb][2 * v] = arow[k]; aF[kb][2 * v + 1] = arow[k + 1];
    }
  }
  v8f acc[8];
#pragma unroll
  for (int t = 0; t < 8; ++t) {
    v8f c = {};
#pragma unroll
    for (int kb = 0; kb < 4; ++kb) {
      v16h bw = *(const v16h*)(WfT + (size_t)(kb * 8 + t) * 512 + lane * 16);
      c = __builtin_amdgcn_wmma_f32_16x16x32_f16(false, aF[kb], false, bw, (short)0, c, false, false);
    }
    acc[t] = c;
  }
  // residual
#pragma unroll
  for (int t = 0; t < 8; ++t)
#pragma unroll
    for (int v = 0; v < 8; ++v)
      acc[t][v] += h[(size_t)(rowBase + v + 8 * hi) * D_ + t * 16 + m];
  // LayerNorm over D=128 per row
  float mu[8], rstd[8];
#pragma unroll
  for (int v = 0; v < 8; ++v) {
    float s = 0.f;
#pragma unroll
    for (int t = 0; t < 8; ++t) s += acc[t][v];
    mu[v] = redSum16(s) * (1.f / D_);
  }
#pragma unroll
  for (int v = 0; v < 8; ++v) {
    float s = 0.f;
#pragma unroll
    for (int t = 0; t < 8; ++t) { float d = acc[t][v] - mu[v]; s += d * d; }
    float var = redSum16(s) * (1.f / D_);
    rstd[v] = rsqrtf(var + LN_EPS);
  }
#pragma unroll
  for (int t = 0; t < 8; ++t) {
    float gv = lng[t * 16 + m], bv = lnb[t * 16 + m];
#pragma unroll
    for (int v = 0; v < 8; ++v) {
      float y = (acc[t][v] - mu[v]) * rstd[v] * gv + bv;
      out[(size_t)(rowBase + v + 8 * hi) * D_ + t * 16 + m] = y;
    }
  }
}

// ------------------------------- launcher -------------------------------
extern "C" void kernel_launch(void* const* d_in, const int* in_sizes, int n_in,
                              void* d_out, int out_size, void* d_ws, size_t ws_size,
                              hipStream_t stream) {
  (void)in_sizes; (void)n_in; (void)out_size; (void)ws_size;
  const float* h   = (const float*)d_in[0];
  const float* rh  = (const float*)d_in[1];
  const float* Wl  = (const float*)d_in[2];
  const float* Wr  = (const float*)d_in[3];
  const float* al  = (const float*)d_in[4];
  const float* ar  = (const float*)d_in[5];
  const float* Wrs = (const float*)d_in[6];
  const float* Wrt = (const float*)d_in[7];
  const float* Wf  = (const float*)d_in[8];
  const float* lng = (const float*)d_in[9];
  const float* lnb = (const float*)d_in[10];
  float* out = (float*)d_out;

  char* ws = (char*)d_ws;
  size_t off = 0;
  auto carve = [&](size_t bytes) -> char* {
    char* p = ws + off;
    off += (bytes + 255) & ~(size_t)255;
    return p;
  };
  _Float16* hh    = (_Float16*)carve((size_t)R_ * D_ * 2);    // 2 MB
  _Float16* wlT   = (_Float16*)carve((size_t)D_ * D_ * 2);    // 32 KB (swizzled)
  _Float16* wrT   = (_Float16*)carve((size_t)D_ * D_ * 2);
  _Float16* wfT   = (_Float16*)carve((size_t)D_ * D_ * 2);
  _Float16* vswz  = (_Float16*)carve((size_t)B_ * H_ * N_ * HD_ * 2); // 2 MB, V-frag tiles
  _Float16* hsa16 = (_Float16*)carve((size_t)R_ * D_ * 2);    // 2 MB
  float*    slb   = (float*)carve((size_t)B_ * H_ * N_ * 4);  // 128 KB
  float*    srb   = (float*)carve((size_t)B_ * H_ * N_ * 4);
  float*    rk    = (float*)carve((size_t)B_ * H_ * N_ * RL_ * 4); // 512 KB
  float*    rq    = (float*)carve((size_t)B_ * H_ * N_ * RL_ * 4);

  // prep
  prep_h_kernel<<<1024, 256, 0, stream>>>(h, hh, R_ * D_);
  prep_w_kernel<<<64, 256, 0, stream>>>(Wl, wlT);
  prep_w_kernel<<<64, 256, 0, stream>>>(Wr, wrT);
  prep_w_kernel<<<64, 256, 0, stream>>>(Wf, wfT);
  prep_r_kernel<<<(B_ * H_ * N_ * RL_ + 255) / 256, 256, 0, stream>>>(rh, Wrs, Wrt, rk, rq);
  zero_kernel<<<64, 256, 0, stream>>>(slb, 2 * B_ * H_ * N_);   // slb,srb contiguous

  // projections: Wl path (scores only), Wr path (scores + V-swizzled fr)
  dim3 gp(R_ / 16, D_ / 16);
  proj_kernel<<<gp, 32, 0, stream>>>(hh, wlT, al, slb, ((_Float16*)0), 0);
  proj_kernel<<<gp, 32, 0, stream>>>(hh, wrT, ar, srb, vswz, 1);

  // flash attention (4 waves/block, one 16-row tile per wave)
  dim3 ga(N_ / 64, H_, B_);
  attn_kernel<<<ga, 128, 0, stream>>>(slb, srb, rk, rq, vswz, hsa16);

  // final projection + residual + LayerNorm
  final_kernel<<<R_ / 16, 32, 0, stream>>>(hsa16, wfT, h, lng, lnb, out);
}
